// TKFA_4303557231352
// MI455X (gfx1250) — compile-verified, matrix operand order checked by reading, not run on hardware
//
#include <hip/hip_runtime.h>
#include <math.h>

// ---------------------------------------------------------------------------
// CDNA5 / gfx1250 implementation of the deformable rank-bucketed attention
// block. All GEMM-shaped work uses v_wmma_f32_16x16x32_f16 (wave32 WMMA).
// ---------------------------------------------------------------------------

typedef __attribute__((ext_vector_type(16))) _Float16 v16h;
typedef __attribute__((ext_vector_type(8)))  float    v8f;

#define HW    4096   // 64*64
#define NKEY  1024   // 32*32
#define BN    2

// rank bucket boundaries (N=1024): N/10, 3N/10, 5N/10, 7N/10
#define K1R 102
#define K3R 307
#define K5R 512
#define K7R 716

// ---------------------------------------------------------------------------
// Generic 1x1-conv GEMM, fully specialized at compile time:
//   out[b][n*M+m] = act( sum_k A[b][k*M+m] * W[n*K+k] + bias[n] )
// A is the channel-concat of A1 (first K1 chans) and A2 (rest) when K1 < K.
// ACT: 0 = none, 1 = prelu(pre[n]), 2 = sigmoid, 3 = tanh
// One wave computes a 32x32 output tile: 2 A-frags x 2 B-frags -> 4 WMMAs per
// K-chunk, giving 2x reuse of every loaded fragment.  All validity predicates
// are lane-uniform and compile-time (K % 32 is 0 or 16), so the inner loop is
// pure loads (constant immediate offsets) + cvt + wmma.
// ---------------------------------------------------------------------------
template<int M, int K, int K1, int ACT>
__global__ void __launch_bounds__(32)
gemm1x1(const float* __restrict__ A1, const float* __restrict__ A2,
        const float* __restrict__ Wt, const float* __restrict__ bias,
        const float* __restrict__ pre, float* __restrict__ out,
        long sA1, long sA2, long sOut)
{
  const int lane = threadIdx.x & 31;
  const int l15 = lane & 15;
  const int m0 = blockIdx.x * 32;
  const int n0 = blockIdx.y * 32;
  const int b  = blockIdx.z;
  const int kb = (lane >= 16) ? 8 : 0;      // per-lane K base inside fragment

  // per-lane base pointers; all further addressing is +constant
  const float* a1 = A1 + (long)b * sA1 + kb * M + m0 + l15;
  const float* a2 = A2 + (long)b * sA2 + kb * M + m0 + l15;
  const float* wp = Wt + (long)(n0 + l15) * K + kb;

  v8f acc[4] = {};
#pragma unroll
  for (int k0 = 0; k0 < K; k0 += 32) {
    const bool full = (k0 + 32 <= K);                 // compile-time
    const float* ab = (k0 < K1) ? (a1 + k0 * M) : (a2 + (k0 - K1) * M);
    v16h af0, af1, bf0, bf1;
#pragma unroll
    for (int e = 0; e < 8; ++e) {                     // K = k0+kb+e   (always valid)
      af0[e] = (_Float16)ab[e * M];
      af1[e] = (_Float16)ab[e * M + 16];
      bf0[e] = (_Float16)wp[k0 + e];
      bf1[e] = (_Float16)wp[16 * K + k0 + e];
    }
#pragma unroll
    for (int e = 8; e < 16; ++e) {                    // K = k0+kb+e+8 (tail: zero)
      af0[e] = full ? (_Float16)ab[(e + 8) * M] : (_Float16)0.f;
      af1[e] = full ? (_Float16)ab[(e + 8) * M + 16] : (_Float16)0.f;
      bf0[e] = full ? (_Float16)wp[k0 + e + 8] : (_Float16)0.f;
      bf1[e] = full ? (_Float16)wp[16 * K + k0 + e + 8] : (_Float16)0.f;
    }
    acc[0] = __builtin_amdgcn_wmma_f32_16x16x32_f16(false, af0, false, bf0,
                                                    (short)0, acc[0], false, false);
    acc[1] = __builtin_amdgcn_wmma_f32_16x16x32_f16(false, af0, false, bf1,
                                                    (short)0, acc[1], false, false);
    acc[2] = __builtin_amdgcn_wmma_f32_16x16x32_f16(false, af1, false, bf0,
                                                    (short)0, acc[2], false, false);
    acc[3] = __builtin_amdgcn_wmma_f32_16x16x32_f16(false, af1, false, bf1,
                                                    (short)0, acc[3], false, false);
  }

#pragma unroll
  for (int tn = 0; tn < 2; ++tn) {
    const int nc = n0 + tn * 16 + l15;
    const float bvv = bias[nc];
    float pv = 0.f;
    if (ACT == 1) pv = pre[nc];
    float* ob = out + (long)b * sOut + (long)nc * M;
#pragma unroll
    for (int tm = 0; tm < 2; ++tm) {
      v8f a = acc[tm * 2 + tn];
#pragma unroll
      for (int r = 0; r < 8; ++r) {
        int m = m0 + tm * 16 + r + ((lane >= 16) ? 8 : 0);
        float v = a[r] + bvv;
        if (ACT == 1)      v = (v >= 0.f) ? v : pv * v;
        else if (ACT == 2) v = 1.f / (1.f + __expf(-v));
        else if (ACT == 3) v = tanhf(v);
        ob[m] = v;
      }
    }
  }
}

// ---------------------------------------------------------------------------
// Depthwise 3x3, stride 2, pad 1 on q_img viewed as (B*HEADS, 80, 64, 64)
// ---------------------------------------------------------------------------
__global__ void dwconv3s2(const float* __restrict__ q, const float* __restrict__ w,
                          const float* __restrict__ b, float* __restrict__ out)
{
  int idx = blockIdx.x * blockDim.x + threadIdx.x;
  if (idx >= 4 * 80 * 1024) return;
  int pos = idx & 1023;
  int ch  = (idx >> 10) % 80;
  int bh  = idx / (80 * 1024);
  int oy = pos >> 5, ox = pos & 31;
  const float* in = q + ((long)(bh >> 1) * 160 + (bh & 1) * 80 + ch) * HW;
  const float* wc = w + ch * 9;
  float s = b[ch];
#pragma unroll
  for (int ky = 0; ky < 3; ++ky) {
    int iy = oy * 2 - 1 + ky;
    if (iy < 0 || iy > 63) continue;
#pragma unroll
    for (int kx = 0; kx < 3; ++kx) {
      int ix = ox * 2 - 1 + kx;
      if (ix < 0 || ix > 63) continue;
      s += in[iy * 64 + ix] * wc[ky * 3 + kx];
    }
  }
  out[idx] = s;
}

// ---------------------------------------------------------------------------
// LayerNorm over 80 channels + SiLU.  One thread per (bh,pos).
// ---------------------------------------------------------------------------
__global__ void ln_silu(const float* __restrict__ t0, const float* __restrict__ g,
                        const float* __restrict__ bta, float* __restrict__ t1)
{
  int idx = blockIdx.x * blockDim.x + threadIdx.x;
  if (idx >= 4 * 1024) return;
  int bh = idx >> 10, pos = idx & 1023;
  const float* p = t0 + (long)bh * 80 * 1024 + pos;
  float mu = 0.f;
  for (int c = 0; c < 80; ++c) mu += p[c * 1024];
  mu *= (1.f / 80.f);
  float var = 0.f;
  for (int c = 0; c < 80; ++c) { float d = p[c * 1024] - mu; var += d * d; }
  var *= (1.f / 80.f);
  float inv = rsqrtf(var + 1e-5f);
  float* o = t1 + (long)bh * 80 * 1024 + pos;
  for (int c = 0; c < 80; ++c) {
    float y = (p[c * 1024] - mu) * inv * g[c] + bta[c];
    o[c * 1024] = y / (1.f + __expf(-y));   // silu
  }
}

// ---------------------------------------------------------------------------
// Pointwise 80->2 offsets + reference grid + clip + bilinear grid_sample of x.
// ---------------------------------------------------------------------------
__global__ void off_sample(const float* __restrict__ t1, const float* __restrict__ wpw,
                           const float* __restrict__ bpw, const float* __restrict__ x,
                           float* __restrict__ sampled)
{
  int idx = blockIdx.x * blockDim.x + threadIdx.x;
  if (idx >= 4 * 1024) return;
  int bh = idx >> 10, pos = idx & 1023;
  const float* p = t1 + (long)bh * 80 * 1024 + pos;
  float o0 = bpw[0], o1 = bpw[1];
  for (int c = 0; c < 80; ++c) {
    float v = p[c * 1024];
    o0 += v * wpw[c];
    o1 += v * wpw[80 + c];
  }
  int oy = pos >> 5, ox = pos & 31;
  float ny = (0.5f + (float)oy) * (2.f / 31.f) - 1.f;
  float nx = (0.5f + (float)ox) * (2.f / 31.f) - 1.f;
  float dpy = fminf(fmaxf(ny + o0, -1.f), 1.f);   // y coord (off ch 0)
  float dpx = fminf(fmaxf(nx + o1, -1.f), 1.f);   // x coord (off ch 1)
  float gx = (dpx + 1.f) * 0.5f * 63.f;
  float gy = (dpy + 1.f) * 0.5f * 63.f;
  float x0 = floorf(gx), y0 = floorf(gy);
  float fx = gx - x0, fy = gy - y0;
  int x0i = min(max((int)x0, 0), 63);
  int x1i = min(max((int)x0 + 1, 0), 63);
  int y0i = min(max((int)y0, 0), 63);
  int y1i = min(max((int)y0 + 1, 0), 63);
  float w00 = (1.f - fx) * (1.f - fy);
  float w01 = fx * (1.f - fy);
  float w10 = (1.f - fx) * fy;
  float w11 = fx * fy;
  int cbase = (bh >> 1) * 160 + (bh & 1) * 80;
  float* outp = sampled + (long)cbase * NKEY + pos;
  for (int d = 0; d < 80; ++d) {
    const float* im = x + (long)(cbase + d) * HW;
    float v = im[y0i * 64 + x0i] * w00 + im[y0i * 64 + x1i] * w01 +
              im[y1i * 64 + x0i] * w10 + im[y1i * 64 + x1i] * w11;
    outp[(long)d * NKEY] = v;
  }
}

// ---------------------------------------------------------------------------
// Fused attention: one block = (b,h, 16 queries).  WMMA q.kT -> LDS dots,
// O(N^2) rank counting -> segments, 5 masked softmaxes, WMMA probs.v.
// Dynamic LDS: dots f32[16][1024] + probs f16[16][1024] + seg u8[16][1024]
// ---------------------------------------------------------------------------
__global__ void __launch_bounds__(256)
attn_kernel(const float* __restrict__ qimg, const float* __restrict__ kv,
            float* __restrict__ obuf)
{
  extern __shared__ char smem[];
  float*         dots  = (float*)smem;                       // 65536 B
  _Float16*      probs = (_Float16*)(smem + 65536);          // 32768 B
  unsigned char* seg   = (unsigned char*)(smem + 98304);     // 16384 B

  const int tid = threadIdx.x;
  const int lane = tid & 31;
  const int w = tid >> 5;
  const int kb = (lane >= 16) ? 8 : 0;
  const int l15 = lane & 15;
  const int q0 = blockIdx.x * 16;
  const int bh = blockIdx.y;
  const int b = bh >> 1, h = bh & 1;
  const float* qb = qimg + (long)b * 160 * HW + (long)h * 80 * HW;
  const float* kp = kv + (long)b * 320 * NKEY + (long)h * 80 * NKEY;
  const float* vp = kv + (long)b * 320 * NKEY + (long)(160 + h * 80) * NKEY;

  // Prefetch the K and V panels for this (b,h) into cache (global_prefetch_b8).
  for (int i = tid; i < (80 * NKEY) / 32; i += 256) {
    __builtin_prefetch(kp + i * 32, 0, 1);
    __builtin_prefetch(vp + i * 32, 0, 1);
  }

  // ---- Phase 1: dots = (q @ kT) * scale via WMMA.  K=80 -> 3 chunks of 32.
  v16h aq[3];
  {
    const float* qbl = qb + kb * HW + q0 + l15;   // +constant per element
#pragma unroll
    for (int c = 0; c < 3; ++c) {
      const bool full = (c < 2);                  // chunk 2: tail (K=80)
#pragma unroll
      for (int e = 0; e < 8; ++e)
        aq[c][e] = (_Float16)qbl[(c * 32 + e) * HW];
#pragma unroll
      for (int e = 8; e < 16; ++e)
        aq[c][e] = full ? (_Float16)qbl[(c * 32 + e + 8) * HW] : (_Float16)0.f;
    }
  }
  const float scale = 0.11180339887498949f;  // 80^-0.5
  const float* kbl = kp + kb * NKEY + l15;
  for (int nt = w; nt < 64; nt += 8) {
    const int n0 = nt * 16;
    v8f acc = {};
#pragma unroll
    for (int c = 0; c < 3; ++c) {
      const bool full = (c < 2);
      v16h bf;
#pragma unroll
      for (int e = 0; e < 8; ++e)
        bf[e] = (_Float16)kbl[(c * 32 + e) * NKEY + n0];
#pragma unroll
      for (int e = 8; e < 16; ++e)
        bf[e] = full ? (_Float16)kbl[(c * 32 + e + 8) * NKEY + n0] : (_Float16)0.f;
      acc = __builtin_amdgcn_wmma_f32_16x16x32_f16(false, aq[c], false, bf,
                                                   (short)0, acc, false, false);
    }
#pragma unroll
    for (int r = 0; r < 8; ++r) {
      int m = r + ((lane >= 16) ? 8 : 0);
      dots[m * 1024 + n0 + l15] = acc[r] * scale;
    }
  }
  __syncthreads();

  // ---- Phase 2: rank counting (argsort ties -> lower index first).
  // Row-serial so the inner LDS read is a broadcast (conflict-free).
  for (int row = 0; row < 16; ++row) {
    float di[4]; int cnt[4];
#pragma unroll
    for (int ii = 0; ii < 4; ++ii) { di[ii] = dots[row * 1024 + tid + 256 * ii]; cnt[ii] = 0; }
    for (int j = 0; j < 1024; ++j) {
      float v = dots[row * 1024 + j];
#pragma unroll
      for (int ii = 0; ii < 4; ++ii) {
        int i = tid + 256 * ii;
        cnt[ii] += (int)((v > di[ii]) || ((v == di[ii]) && (j < i)));
      }
    }
#pragma unroll
    for (int ii = 0; ii < 4; ++ii) {
      int rk = cnt[ii];
      unsigned char s = (rk < K1R) ? 0 : (rk < K3R) ? 1 : (rk < K5R) ? 2
                       : (rk < K7R) ? 3 : 4;
      seg[row * 1024 + tid + 256 * ii] = s;
    }
  }
  __syncthreads();

  // ---- Phase 3/4 per branch: masked softmax -> f16 probs, WMMA probs @ v.
  const float* vbl0 = vp + (long)l15 * NKEY + kb;      // + w*16*NKEY for dd tile
  const _Float16* pbl = probs + l15 * 1024 + kb;
  for (int br = 0; br < 5; ++br) {
#pragma unroll 1
    for (int rr = 0; rr < 2; ++rr) {
      int row = w * 2 + rr;               // 8 waves x 2 rows = 16
      float mx = -3.4e38f;
      for (int i = 0; i < 32; ++i) {
        int n = lane + 32 * i;
        bool msk = (br == 0) || (seg[row * 1024 + n] == (unsigned char)(br - 1));
        if (msk) mx = fmaxf(mx, dots[row * 1024 + n]);
      }
      for (int off = 16; off >= 1; off >>= 1) mx = fmaxf(mx, __shfl_xor(mx, off));
      float sum = 0.f;
      for (int i = 0; i < 32; ++i) {
        int n = lane + 32 * i;
        bool msk = (br == 0) || (seg[row * 1024 + n] == (unsigned char)(br - 1));
        float e = msk ? __expf(dots[row * 1024 + n] - mx) : 0.f;
        probs[row * 1024 + n] = (_Float16)e;
        sum += e;
      }
      for (int off = 16; off >= 1; off >>= 1) sum += __shfl_xor(sum, off);
      float inv = 1.f / sum;
      for (int i = 0; i < 32; ++i) {
        int n = lane + 32 * i;
        probs[row * 1024 + n] = (_Float16)((float)probs[row * 1024 + n] * inv);
      }
    }
    __syncthreads();

    if (w < 5) {                           // 5 waves x 16 dims = 80 head dims
      const int ddb = w * 16;
      const float* vbl = vbl0 + (long)ddb * NKEY;
      v8f acc = {};
#pragma unroll 4
      for (int c = 0; c < 32; ++c) {
        v16h af, bf;
#pragma unroll
        for (int e = 0; e < 8; ++e) {
          af[e]     = pbl[c * 32 + e];
          af[e + 8] = pbl[c * 32 + 16 + e];
          bf[e]     = (_Float16)vbl[c * 32 + e];
          bf[e + 8] = (_Float16)vbl[c * 32 + 16 + e];
        }
        acc = __builtin_amdgcn_wmma_f32_16x16x32_f16(false, af, false, bf,
                                                     (short)0, acc, false, false);
      }
      float* ob = obuf + (long)br * (2L * 160 * HW) + (long)b * 160 * HW +
                  (long)(h * 80 + ddb + l15) * HW;
#pragma unroll
      for (int r = 0; r < 8; ++r) {
        int m = q0 + r + ((lane >= 16) ? 8 : 0);
        ob[m] = acc[r];
      }
    }
    __syncthreads();
  }
}

// ---------------------------------------------------------------------------
// Depthwise 11x11 pad 5 on t[:, choff:choff+80, :, :]
// ---------------------------------------------------------------------------
__global__ void dwconv11(const float* __restrict__ tin, int choff,
                         const float* __restrict__ w, const float* __restrict__ bias,
                         float* __restrict__ out)
{
  int idx = blockIdx.x * blockDim.x + threadIdx.x;
  if (idx >= BN * 80 * HW) return;
  int pos = idx & 4095;
  int ch  = (idx >> 12) % 80;
  int b   = idx / (80 * HW);
  int y = pos >> 6, x = pos & 63;
  const float* in = tin + ((long)b * 160 + choff + ch) * HW;
  const float* wc = w + ch * 121;
  float s = bias[ch];
  for (int ky = 0; ky < 11; ++ky) {
    int iy = y + ky - 5;
    if (iy < 0 || iy > 63) continue;
    for (int kx = 0; kx < 11; ++kx) {
      int ix = x + kx - 5;
      if (ix < 0 || ix > 63) continue;
      s += in[iy * 64 + ix] * wc[ky * 11 + kx];
    }
  }
  out[((long)b * 80 + ch) * HW + pos] = s;
}

// ---------------------------------------------------------------------------
// acc (+)= aj * ( Z*h + (1-Z)*(R*xx) )
// ---------------------------------------------------------------------------
__global__ void combine(const float* __restrict__ Z, const float* __restrict__ R,
                        const float* __restrict__ h, const float* __restrict__ xx,
                        const float* __restrict__ aj, float* __restrict__ acc, int first)
{
  long i = (long)blockIdx.x * blockDim.x + threadIdx.x;
  if (i >= 2L * 160 * HW) return;
  float z = Z[i];
  float v = z * h[i] + (1.f - z) * (R[i] * xx[i]);
  float a = aj[0];
  acc[i] = first ? (v * a) : (acc[i] + v * a);
}

// ---------------------------------------------------------------------------
extern "C" void kernel_launch(void* const* d_in, const int* in_sizes, int n_in,
                              void* d_out, int out_size, void* d_ws, size_t ws_size,
                              hipStream_t stream)
{
  (void)in_sizes; (void)n_in; (void)out_size; (void)ws_size;
  const float* x     = (const float*)d_in[0];
  const float* wq    = (const float*)d_in[1];
  const float* bq    = (const float*)d_in[2];
  const float* wkv   = (const float*)d_in[3];
  const float* bkv   = (const float*)d_in[4];
  const float* wo_dw = (const float*)d_in[5];
  const float* bo_dw = (const float*)d_in[6];
  const float* g_ln  = (const float*)d_in[7];
  const float* b_ln  = (const float*)d_in[8];
  const float* wo_pw = (const float*)d_in[9];
  const float* bo_pw = (const float*)d_in[10];
  const float* wm    = (const float*)d_in[11];
  const float* bm    = (const float*)d_in[12];
  const float* am    = (const float*)d_in[13];
  const float* wz_dw = (const float*)d_in[14];
  const float* bz_dw = (const float*)d_in[15];
  const float* wz_pw = (const float*)d_in[16];
  const float* bz_pw = (const float*)d_in[17];
  const float* wr_dw = (const float*)d_in[18];
  const float* br_dw = (const float*)d_in[19];
  const float* wr_pw = (const float*)d_in[20];
  const float* br_pw = (const float*)d_in[21];
  const float* wout  = (const float*)d_in[26];
  const float* bout  = (const float*)d_in[27];

  float* ws      = (float*)d_ws;
  float* q_img   = ws;                       // 2*160*4096
  float* t0      = q_img + 1310720;          // 4*80*1024
  float* t1      = t0 + 327680;              // 4*80*1024
  float* sampled = t1 + 327680;              // 2*160*1024
  float* kvb     = sampled + 327680;         // 2*320*1024
  float* obuf    = kvb + 655360;             // 5 * 2*160*4096
  float* tmod    = obuf + 5L * 1310720;      // 2*160*4096
  float* zdw     = tmod + 1310720;           // 2*80*4096
  float* Zb      = zdw + 655360;             // 2*160*4096
  float* rdw     = Zb + 1310720;             // 2*80*4096
  float* Rb      = rdw + 655360;             // 2*160*4096
  float* accb    = Rb + 1310720;             // 2*160*4096

  // q = conv1x1(x)  [K=160 -> N=160, M=4096]
  gemm1x1<HW, 160, 160, 0><<<dim3(128, 5, 2), 32, 0, stream>>>(
      x, x, wq, bq, nullptr, q_img, 160L * HW, 0, 160L * HW);

  // offset branch
  dwconv3s2<<<1280, 256, 0, stream>>>(q_img, wo_dw, bo_dw, t0);
  ln_silu<<<16, 256, 0, stream>>>(t0, g_ln, b_ln, t1);
  off_sample<<<16, 256, 0, stream>>>(t1, wo_pw, bo_pw, x, sampled);

  // kv = conv1x1(sampled)  [K=160 -> N=320, M=1024]
  gemm1x1<NKEY, 160, 160, 0><<<dim3(32, 10, 2), 32, 0, stream>>>(
      sampled, sampled, wkv, bkv, nullptr, kvb, 160L * NKEY, 0, 320L * NKEY);

  // fused attention -> o0..o4
  hipFuncSetAttribute(reinterpret_cast<const void*>(attn_kernel),
                      hipFuncAttributeMaxDynamicSharedMemorySize, 114688);
  attn_kernel<<<dim3(256, 4), 256, 114688, stream>>>(q_img, kvb, obuf);

  // 4 modulator blocks + weighted accumulation
  for (int j = 1; j <= 4; ++j) {
    const float* oj = obuf + (long)j * 2 * 160 * HW;
    const float* aj = (const float*)d_in[21 + j];   // a1..a4 at indices 22..25
    // t = prelu(conv1x1(cat[o0, oj]))  [K=320 (160+160) -> N=160]
    gemm1x1<HW, 320, 160, 1><<<dim3(128, 5, 2), 32, 0, stream>>>(
        obuf, oj, wm, bm, am, tmod, 160L * HW, 160L * HW, 160L * HW);
    // Z = sigmoid(pw(dw(U)))
    dwconv11<<<2560, 256, 0, stream>>>(tmod, 0, wz_dw, bz_dw, zdw);
    gemm1x1<HW, 80, 80, 2><<<dim3(128, 5, 2), 32, 0, stream>>>(
        zdw, zdw, wz_pw, bz_pw, nullptr, Zb, 80L * HW, 0, 160L * HW);
    // R = tanh(pw(dw(M)))
    dwconv11<<<2560, 256, 0, stream>>>(tmod, 80, wr_dw, br_dw, rdw);
    gemm1x1<HW, 80, 80, 3><<<dim3(128, 5, 2), 32, 0, stream>>>(
        rdw, rdw, wr_pw, br_pw, nullptr, Rb, 80L * HW, 0, 160L * HW);
    combine<<<5120, 256, 0, stream>>>(Zb, Rb, obuf, oj, aj, accb, j == 1);
  }

  // final conv1x1 -> d_out
  gemm1x1<HW, 160, 160, 0><<<dim3(128, 5, 2), 32, 0, stream>>>(
      accb, accb, wout, bout, nullptr, (float*)d_out, 160L * HW, 0, 160L * HW);
}
